// GaussParzenExtractor_50629074485729
// MI455X (gfx1250) — matrix operand order repaired and checked by reading, not compiled, not run on hardware
//
#include <hip/hip_runtime.h>
#include <hip/hip_bf16.h>
#include <stdint.h>

#define PBINS 32
#define NV    1024
#define BB    4
#define HH    160
#define WW    160
#define NPIX  (BB*HH*WW)
#define SIGMA 0.05f

typedef __attribute__((ext_vector_type(16))) _Float16      v16h;
typedef __attribute__((ext_vector_type(8)))  float         v8f;
typedef __attribute__((ext_vector_type(8)))  unsigned int  v8u;

union Frag { v8u u; v16h h; };

// ---------------- pass 0: zero segment counters ----------------
__global__ void init_counts_kernel(uint32_t* __restrict__ cnt) {
  int t = blockIdx.x * blockDim.x + threadIdx.x;
  if (t < NV) cnt[t] = 0u;
}

// ---------------- pass 1: per-segment pixel counts ----------------
__global__ void count_kernel(const int* __restrict__ seg, uint32_t* __restrict__ cnt) {
  int n = blockIdx.x * blockDim.x + threadIdx.x;
  if (n < NPIX) atomicAdd(&cnt[seg[n]], 1u);
}

// ---------------- pass 2: exclusive prefix scan over 1024 counts ----------------
__global__ void scan_kernel(const uint32_t* __restrict__ cnt,
                            uint32_t* __restrict__ off,
                            uint32_t* __restrict__ cur) {
  __shared__ uint32_t sh[NV];
  int t = threadIdx.x;
  uint32_t c = cnt[t];
  sh[t] = c;
  __syncthreads();
  #pragma unroll
  for (int o = 1; o < NV; o <<= 1) {
    uint32_t v = (t >= o) ? sh[t - o] : 0u;
    __syncthreads();
    sh[t] += v;
    __syncthreads();
  }
  uint32_t excl = sh[t] - c;   // exclusive prefix
  off[t] = excl;
  cur[t] = excl;
}

// ---------------- pass 3: scatter pixel indices into segment-sorted order ----------------
__global__ void scatter_kernel(const int* __restrict__ seg,
                               uint32_t* __restrict__ cur,
                               uint32_t* __restrict__ idx) {
  int n = blockIdx.x * blockDim.x + threadIdx.x;
  if (n < NPIX) {
    uint32_t p = atomicAdd(&cur[seg[n]], 1u);
    idx[p] = (uint32_t)n;
  }
}

// ---------------- pass 4: per-segment 32x32 Parzen joint histogram via WMMA ----------------
// One wave (32 lanes) per segment, templated on pair (0: coords, 1: grads).
// For each chunk of 32 pixels:
//   hst(32x32) += Wa^T (32bins x 32px, f16) * Wb (32px x 32bins, f16)
// as 4 x v_wmma_f32_16x16x32_f16 tiles with f32 accumulators resident in VGPRs.
// The tail guard is branchless (index clamp + valid multiplier) so EXEC stays
// all-ones through the whole loop: no saveexec/restore around the gathers.
template <int PAIR>
__global__ __launch_bounds__(32) void hist_wmma_kernel(
    const int*      __restrict__ byx,
    const float*    __restrict__ grad,
    const uint32_t* __restrict__ cnt,
    const uint32_t* __restrict__ off,
    const uint32_t* __restrict__ idx,
    float*          __restrict__ out) {

  const int s    = blockIdx.x;   // segment id
  const int lane = threadIdx.x;

  // LDS staging, [bin][pixel] so K (pixel) is the contiguous axis for both
  // the A fragment (packed K pairs per row M) and the B fragment.
  __shared__ _Float16 Wa[PBINS][PBINS];
  __shared__ _Float16 Wb[PBINS][PBINS];

  const uint32_t c0 = off[s];
  const uint32_t cN = cnt[s];

  v8f acc00 = {}, acc01 = {}, acc10 = {}, acc11 = {};

  const int M  = lane & 15;   // A row / B col / C col within 16-tile
  const int hi = lane >> 4;

  for (uint32_t base = 0; base < cN; base += 32) {
    // prefetch next chunk's sorted indices (global_prefetch_b8)
    __builtin_prefetch(idx + c0 + base + 32, 0, 1);

    // ---- branchless gather: clamp tail lanes to a valid slot, zero later ----
    uint32_t i     = base + (uint32_t)lane;
    uint32_t iw    = (i < cN) ? i : (cN - 1u);   // cN >= 1 inside the loop
    float    valid = (i < cN) ? 1.f : 0.f;
    uint32_t n     = idx[c0 + iw];

    float va, vb;
    if (PAIR == 0) {
      int y = byx[NPIX + n];
      int x = byx[2 * NPIX + n];
      va = 2.f * (float)y / (float)HH - 1.f;
      vb = 2.f * (float)x / (float)WW - 1.f;
    } else {
      uint32_t b = n / (HH * WW);
      uint32_t r = n - b * (HH * WW);
      va = grad[(b * 2 + 0) * (HH * WW) + r];
      vb = grad[(b * 2 + 1) * (HH * WW) + r];
    }

    #pragma unroll
    for (int p = 0; p < PBINS; ++p) {
      float cbin = 2.f * ((float)p + 0.5f) / (float)PBINS - 1.f;
      float ta = (va - cbin) * (1.f / SIGMA);
      float tb = (vb - cbin) * (1.f / SIGMA);
      // exp(-0.5 t^2) = exp2(-0.5*log2(e) * t^2); v_exp_f32 co-executes with WMMA
      float wa = __builtin_amdgcn_exp2f(-0.72134752f * ta * ta) * valid;
      float wb = __builtin_amdgcn_exp2f(-0.72134752f * tb * tb) * valid;
      Wa[p][lane] = (_Float16)wa;
      Wb[p][lane] = (_Float16)wb;
    }
    __syncthreads();

    // ---- load WMMA fragments per the CDNA5 16-bit A(16x32)/B(32x16) layouts ----
    const uint32_t* pa = (const uint32_t*)&Wa[0][0];   // 16 dwords per bin row
    const uint32_t* pb = (const uint32_t*)&Wb[0][0];
    Frag aLo, aHi, bLo, bHi;
    #pragma unroll
    for (int v = 0; v < 8; ++v) {
      // A: lane rows M=L%16; VGPR v holds K pair: v<4 -> K=8*hi+2v, else 16+8*hi+2(v-4)
      int kA = (v < 4) ? (8 * hi + 2 * v) : (16 + 8 * hi + 2 * (v - 4));
      // B: lane col N=L%16; VGPR v holds K pair: v<4 -> K=8*hi+2v, else 16+8*hi+2(v-4)
      int kB = (v < 4) ? (8 * hi + 2 * v) : (16 + 8 * hi + 2 * (v - 4));
      aLo.u[v] = pa[(0  + M) * (PBINS / 2) + (kA >> 1)];
      aHi.u[v] = pa[(16 + M) * (PBINS / 2) + (kA >> 1)];
      bLo.u[v] = pb[(0  + M) * (PBINS / 2) + (kB >> 1)];
      bHi.u[v] = pb[(16 + M) * (PBINS / 2) + (kB >> 1)];
    }
    __syncthreads();   // reads done before next chunk overwrites LDS

    acc00 = __builtin_amdgcn_wmma_f32_16x16x32_f16(false, aLo.h, false, bLo.h,
                                                   (short)0, acc00, false, false);
    acc01 = __builtin_amdgcn_wmma_f32_16x16x32_f16(false, aLo.h, false, bHi.h,
                                                   (short)0, acc01, false, false);
    acc10 = __builtin_amdgcn_wmma_f32_16x16x32_f16(false, aHi.h, false, bLo.h,
                                                   (short)0, acc10, false, false);
    acc11 = __builtin_amdgcn_wmma_f32_16x16x32_f16(false, aHi.h, false, bHi.h,
                                                   (short)0, acc11, false, false);
  }

  // ---- normalize by segment size (den = sizes * (P/32)^2 = sizes) and store ----
  float inv = (cN > 0) ? (1.f / (float)cN) : 0.f;
  float* o = out + (size_t)(s * 2 + PAIR) * (PBINS * PBINS);
  // C layout: VGPR r, lanes 0-15: row M=r, col=lane; lanes 16-31: row M=r+8, col=lane-16
  #pragma unroll
  for (int r = 0; r < 8; ++r) {
    int rowL = r + 8 * hi;
    o[(0  + rowL) * PBINS + (0  + M)] = acc00[r] * inv;
    o[(0  + rowL) * PBINS + (16 + M)] = acc01[r] * inv;
    o[(16 + rowL) * PBINS + (0  + M)] = acc10[r] * inv;
    o[(16 + rowL) * PBINS + (16 + M)] = acc11[r] * inv;
  }
}

extern "C" void kernel_launch(void* const* d_in, const int* in_sizes, int n_in,
                              void* d_out, int out_size, void* d_ws, size_t ws_size,
                              hipStream_t stream) {
  const int*   seg  = (const int*)d_in[0];
  const int*   byx  = (const int*)d_in[1];
  const float* grad = (const float*)d_in[2];
  // d_in[3] = nV device scalar; NV=1024 is fixed by the reference problem.
  float* out = (float*)d_out;

  uint32_t* cnt = (uint32_t*)d_ws;   // [NV]
  uint32_t* off = cnt + NV;          // [NV]
  uint32_t* cur = off + NV;          // [NV]
  uint32_t* idx = cur + NV;          // [NPIX]

  init_counts_kernel<<<(NV + 255) / 256, 256, 0, stream>>>(cnt);
  count_kernel<<<(NPIX + 255) / 256, 256, 0, stream>>>(seg, cnt);
  scan_kernel<<<1, NV, 0, stream>>>(cnt, off, cur);
  scatter_kernel<<<(NPIX + 255) / 256, 256, 0, stream>>>(seg, cur, idx);

  hist_wmma_kernel<0><<<NV, 32, 0, stream>>>(byx, grad, cnt, off, idx, out);
  hist_wmma_kernel<1><<<NV, 32, 0, stream>>>(byx, grad, cnt, off, idx, out);
}
